// SelfAttention_84628035600548
// MI455X (gfx1250) — compile-verified, matrix operand order checked by reading, not run on hardware
//
#include <hip/hip_runtime.h>
#include <hip/hip_bf16.h>

// ---------------------------------------------------------------------------
// Self-attention (full-dim, no head split) for MI455X / gfx1250.
// All GEMM stages -> v_wmma_f32_16x16x32_bf16 (wave32 WMMA). B panels staged
// to LDS in 64x128 chunks by the Tensor Data Mover (tensor_load_to_lds with
// HW LDS padding for conflict-free banking), double buffered, completion via
// s_wait_tensorcnt. V is transposed for free in the qkv GEMM epilogue so P@V
// is a K-contiguous GEMM too.
// Pipeline: cvt(f32->bf16) -> GEMM qkv (writes Q|K and V^T) ->
//           GEMM scores(*0.125) -> softmax -> GEMM P@V^T -> GEMM out(+bias).
// Workspace: ~362 MB.
// ---------------------------------------------------------------------------

typedef __attribute__((ext_vector_type(8)))  __bf16 bf16x8;
typedef __attribute__((ext_vector_type(16))) __bf16 v16bf;
typedef __attribute__((ext_vector_type(8)))  float  v8f;

typedef __attribute__((ext_vector_type(4))) unsigned int tdm4u;
typedef __attribute__((ext_vector_type(8))) int          tdm8i;
typedef __attribute__((ext_vector_type(4))) int          tdm4i;

#define DIM    1024
#define SEQ    2048
#define BATCH  8
#define ROWS   (BATCH * SEQ)     // 16384
#define QK_LD  (2 * DIM)         // 2048 (q|k buffer)
#define SCALE  0.125f            // (DIM/NUM_HEADS)^-0.5 = 64^-0.5

// B chunk staged per TDM op: 64 N-rows x 128 K bf16. HW pad of 4 DWORDs
// (8 bf16) after each 64-DWORD (128 bf16) row -> LDS row stride 136 elements
// (68 DWORDs; 68*r mod 64 = 4r -> conflict-free b128 reads across 16 rows).
#define KCHUNK 128
#define CROW   136
#define CBUF   (64 * CROW)       // elements per LDS buffer (17408 B)

#if __has_builtin(__builtin_amdgcn_tensor_load_to_lds) && \
    __has_builtin(__builtin_amdgcn_s_wait_tensorcnt)
#define USE_TDM 1
#else
#define USE_TDM 0
#endif

// --- WMMA helper ------------------------------------------------------------
__device__ __forceinline__ v8f wmma_bf16(v16bf a, v16bf b, v8f c) {
  return __builtin_amdgcn_wmma_f32_16x16x32_bf16(false, a, false, b,
                                                 (short)0, c, false, false);
}

// Fragment from row-major [rows, K] bf16, K contiguous (global memory).
// ISA 16-bit 16x32 layout: lanes 0-15 -> row=lane, K {0..7,16..23};
// lanes 16-31 -> same rows, K {8..15,24..31}. Two b128 loads per lane.
__device__ __forceinline__ v16bf frag_ld(const __bf16* __restrict__ base,
                                         int ld, int r0, int k0) {
  const int lane = threadIdx.x & 31;
  const __bf16* p = base + (size_t)(r0 + (lane & 15)) * ld + k0 +
                    ((lane >> 4) << 3);
  bf16x8 lo = *(const bf16x8*)(p);
  bf16x8 hi = *(const bf16x8*)(p + 16);
  return __builtin_shufflevector(lo, hi, 0, 1, 2, 3, 4, 5, 6, 7,
                                         8, 9, 10, 11, 12, 13, 14, 15);
}

// Same fragment out of the LDS-staged B chunk (row stride CROW), K offset kk.
__device__ __forceinline__ v16bf frag_ld_lds(const __bf16* tile, int n_base,
                                             int kk) {
  const int lane = threadIdx.x & 31;
  const __bf16* p = tile + (n_base + (lane & 15)) * CROW + kk +
                    ((lane >> 4) << 3);
  bf16x8 lo = *(const bf16x8*)(p);
  bf16x8 hi = *(const bf16x8*)(p + 16);
  return __builtin_shufflevector(lo, hi, 0, 1, 2, 3, 4, 5, 6, 7,
                                         8, 9, 10, 11, 12, 13, 14, 15);
}

#if USE_TDM
// TDM: stage a 64x128 bf16 tile (row stride ldb) into LDS at lds_byte_off,
// with HW padding of 4 DWORDs after every 64 DWORDs so the LDS image has row
// stride CROW=136 elements. D# layout per CDNA5 ISA 8.3/8.4.
__device__ __forceinline__ void tdm_stage_b_chunk(const __bf16* gsrc, int ldb,
                                                  int brows,
                                                  unsigned lds_byte_off) {
  const unsigned long long ga = (unsigned long long)(size_t)gsrc;
  tdm4u g0;
  g0[0] = 1u;                                              // count=1 (valid)
  g0[1] = lds_byte_off;                                    // lds_addr
  g0[2] = (unsigned)ga;                                    // global_addr[31:0]
  g0[3] = (unsigned)((ga >> 32) & 0x01FFFFFFu) | (2u << 30); // [56:32]|type=2
  const unsigned dim0 = (unsigned)ldb;                     // tensor_dim0
  const unsigned dim1 = (unsigned)brows;                   // tensor_dim1
  tdm8i g1;
  g1[0] = (int)((1u << 16)        // data_size = 2 bytes
              | (1u << 20)        // pad_enable
              | (5u << 22)        // pad_interval = 64 DWORDs (128 bf16)
              | (3u << 25));      // pad_amount   = 4 DWORDs  (8 bf16)
  g1[1] = (int)((dim0 & 0xFFFFu) << 16);                   // dim0[15:0]
  g1[2] = (int)(((dim0 >> 16) & 0xFFFFu) | ((dim1 & 0xFFFFu) << 16));
  g1[3] = (int)(((dim1 >> 16) & 0xFFFFu) | ((unsigned)KCHUNK << 16)); // tile_dim0
  g1[4] = 64;                                              // tile_dim1 = 64
  g1[5] = (int)(unsigned)ldb;                              // dim0_stride lo
  g1[6] = 0;
  g1[7] = 0;
  tdm4i g2 = {0, 0, 0, 0};
  tdm4i g3 = {0, 0, 0, 0};
  tdm8i g4 = {0, 0, 0, 0, 0, 0, 0, 0};
  __builtin_amdgcn_tensor_load_to_lds(g0, g1, g2, g3, g4, 0);
}
#else
// Fallback: cooperative 64x128 chunk copy, 4 b128 loads/stores per thread.
__device__ __forceinline__ void stage_chunk_fallback(const __bf16* B, int ldb,
                                                     int n0, int kc,
                                                     __bf16* dst) {
  const int row = threadIdx.x >> 2;
  const int c0  = (threadIdx.x & 3) << 3;
  const __bf16* src = B + (size_t)(n0 + row) * ldb + kc;
#pragma unroll
  for (int j = 0; j < KCHUNK; j += 32)
    *(bf16x8*)(dst + row * CROW + c0 + j) = *(const bf16x8*)(src + c0 + j);
}
#endif

// Block = 256 threads (8 waves). Block tile: 256 M x 64 N. Wave: 32 M x 64 N.
// B chunk (64x128) staged to LDS, double buffered, 32 WMMAs per chunk.
// acc[mi*4+ni] accumulates C(16x16) at (m0 + 16*mi, n0 + 16*ni).
__device__ __forceinline__ void gemm_compute(const __bf16* __restrict__ A, int lda,
                                             const __bf16* __restrict__ B, int ldb,
                                             int brows, int K, int m0, int n0,
                                             __bf16* lds, v8f acc[8]) {
#if USE_TDM
  const int wv = threadIdx.x >> 5;
  const unsigned lds_base = (unsigned)(size_t)lds;
  if (wv == 0)
    tdm_stage_b_chunk(B + (size_t)n0 * ldb, ldb, brows, lds_base);
  __builtin_amdgcn_s_wait_tensorcnt(0);   // no-op for non-issuing waves
  __syncthreads();
#else
  stage_chunk_fallback(B, ldb, n0, 0, lds);
  __syncthreads();
#endif

  for (int kc = 0; kc < K; kc += KCHUNK) {
    const int buf = (kc / KCHUNK) & 1;
    const __bf16* tile = lds + buf * CBUF;
    const bool have_next = (kc + KCHUNK < K);
#if USE_TDM
    if (have_next && wv == 0)
      tdm_stage_b_chunk(B + (size_t)n0 * ldb + (kc + KCHUNK), ldb, brows,
                        lds_base + (unsigned)((buf ^ 1) * CBUF * 2));
#endif
    if (have_next)
      __builtin_prefetch(A + (size_t)(m0 + (threadIdx.x & 15)) * lda + kc +
                             KCHUNK, 0, 1);
#pragma unroll
    for (int kk = 0; kk < KCHUNK; kk += 32) {
      const int k0 = kc + kk;
      v16bf a0 = frag_ld(A, lda, m0, k0);
      v16bf a1 = frag_ld(A, lda, m0 + 16, k0);
      v16bf b0 = frag_ld_lds(tile, 0,  kk);
      v16bf b1 = frag_ld_lds(tile, 16, kk);
      v16bf b2 = frag_ld_lds(tile, 32, kk);
      v16bf b3 = frag_ld_lds(tile, 48, kk);
      acc[0] = wmma_bf16(a0, b0, acc[0]);
      acc[4] = wmma_bf16(a1, b0, acc[4]);
      acc[1] = wmma_bf16(a0, b1, acc[1]);
      acc[5] = wmma_bf16(a1, b1, acc[5]);
      acc[2] = wmma_bf16(a0, b2, acc[2]);
      acc[6] = wmma_bf16(a1, b2, acc[6]);
      acc[3] = wmma_bf16(a0, b3, acc[3]);
      acc[7] = wmma_bf16(a1, b3, acc[7]);
    }
    if (have_next) {
#if USE_TDM
      __builtin_amdgcn_s_wait_tensorcnt(0);
      __syncthreads();
#else
      __syncthreads();
      stage_chunk_fallback(B, ldb, n0, kc + KCHUNK, lds + (buf ^ 1) * CBUF);
      __syncthreads();
#endif
    }
  }
}

// --- fp32 -> bf16 convert ---------------------------------------------------
__global__ void __launch_bounds__(256)
cvt_bf16_kernel(const float* __restrict__ src, __bf16* __restrict__ dst, int n) {
  int i = (blockIdx.x * blockDim.x + threadIdx.x) * 4;
  if (i < n) {
    float4 v = *(const float4*)(src + i);
    dst[i + 0] = (__bf16)v.x;
    dst[i + 1] = (__bf16)v.y;
    dst[i + 2] = (__bf16)v.z;
    dst[i + 3] = (__bf16)v.w;
  }
}

// --- GEMM 1: qkv projection. Q,K -> qk[ROWS, 2048]; V -> vt[BATCH, DIM, SEQ].
__global__ void __launch_bounds__(256)
gemm_qkv_kernel(const __bf16* __restrict__ A,     // x_bf16 [ROWS, DIM]
                const __bf16* __restrict__ B,     // Wqkv_bf16 [3*DIM, DIM]
                __bf16* __restrict__ QK,          // [ROWS, 2*DIM]
                __bf16* __restrict__ VT) {        // [BATCH, DIM, SEQ]
  __shared__ __align__(16) __bf16 lds[2 * CBUF];
  const int wave = threadIdx.x >> 5;
  const int lane = threadIdx.x & 31;
  const int m0 = blockIdx.y * 256 + wave * 32;
  const int n0 = blockIdx.x * 64;                 // 0..3071
  v8f acc[8] = {};
  gemm_compute(A, DIM, B, DIM, 3 * DIM, DIM, m0, n0, lds, acc);

  const int nc = lane & 15;
  if (n0 < 2 * DIM) {                             // Q|K region: direct store
#pragma unroll
    for (int mi = 0; mi < 2; ++mi) {
      const int mb = m0 + 16 * mi + ((lane >> 4) << 3);
#pragma unroll
      for (int r = 0; r < 8; ++r) {
        __bf16* crow = QK + (size_t)(mb + r) * QK_LD + n0 + nc;
#pragma unroll
        for (int ni = 0; ni < 4; ++ni) crow[16 * ni] = (__bf16)acc[mi * 4 + ni][r];
      }
    }
  } else {                                        // V region: transposed store
    const int bz = (blockIdx.y * 256) >> 11;      // batch (uniform per block)
    const int sb = (m0 & (SEQ - 1)) + ((lane >> 4) << 3);
#pragma unroll
    for (int mi = 0; mi < 2; ++mi) {
#pragma unroll
      for (int ni = 0; ni < 4; ++ni) {
        const int d = n0 + 16 * ni + nc - 2 * DIM;
        bf16x8 vv;
#pragma unroll
        for (int r = 0; r < 8; ++r) vv[r] = (__bf16)acc[mi * 4 + ni][r];
        *(bf16x8*)(VT + ((size_t)bz * DIM + d) * SEQ + sb + 16 * mi) = vv;
      }
    }
  }
}

// --- GEMM 2: S = (q @ k^T) * SCALE, f32 [BATCH, SEQ, SEQ] -------------------
__global__ void __launch_bounds__(256)
gemm_scores_kernel(const __bf16* __restrict__ QK, float* __restrict__ S) {
  __shared__ __align__(16) __bf16 lds[2 * CBUF];
  const int bz = blockIdx.z;
  const __bf16* Q  = QK + (size_t)bz * SEQ * QK_LD;       // cols 0..1023
  const __bf16* Kp = Q + DIM;                             // cols 1024..2047
  float* Sb = S + (size_t)bz * SEQ * SEQ;
  const int wave = threadIdx.x >> 5;
  const int lane = threadIdx.x & 31;
  const int m0 = blockIdx.y * 256 + wave * 32;
  const int n0 = blockIdx.x * 64;
  v8f acc[8] = {};
  gemm_compute(Q, QK_LD, Kp, QK_LD, SEQ, DIM, m0, n0, lds, acc);

  const int nc = n0 + (lane & 15);
#pragma unroll
  for (int mi = 0; mi < 2; ++mi) {
    const int mb = m0 + 16 * mi + ((lane >> 4) << 3);
#pragma unroll
    for (int r = 0; r < 8; ++r) {
      float* srow = Sb + (size_t)(mb + r) * SEQ + nc;
#pragma unroll
      for (int ni = 0; ni < 4; ++ni) srow[16 * ni] = acc[mi * 4 + ni][r] * SCALE;
    }
  }
}

// --- Softmax: one block per row, 2048 cols, bf16 out ------------------------
__global__ void __launch_bounds__(256)
softmax_kernel(const float* __restrict__ S, __bf16* __restrict__ P) {
  const int row  = blockIdx.x;
  const int tid  = threadIdx.x;
  const int wave = tid >> 5;
  const int lane = tid & 31;
  const float* s = S + (size_t)row * SEQ + tid * 8;
  float4 v0 = *(const float4*)(s);
  float4 v1 = *(const float4*)(s + 4);
  float x[8] = {v0.x, v0.y, v0.z, v0.w, v1.x, v1.y, v1.z, v1.w};

  float m = x[0];
#pragma unroll
  for (int i = 1; i < 8; ++i) m = fmaxf(m, x[i]);
#pragma unroll
  for (int off = 16; off >= 1; off >>= 1) m = fmaxf(m, __shfl_xor(m, off, 32));

  __shared__ float smax[8];
  __shared__ float ssum[8];
  if (lane == 0) smax[wave] = m;
  __syncthreads();
#pragma unroll
  for (int w = 0; w < 8; ++w) m = fmaxf(m, smax[w]);

  float sum = 0.f;
#pragma unroll
  for (int i = 0; i < 8; ++i) { x[i] = __expf(x[i] - m); sum += x[i]; }
#pragma unroll
  for (int off = 16; off >= 1; off >>= 1) sum += __shfl_xor(sum, off, 32);
  if (lane == 0) ssum[wave] = sum;
  __syncthreads();
  sum = 0.f;
#pragma unroll
  for (int w = 0; w < 8; ++w) sum += ssum[w];

  const float inv = 1.0f / sum;
  __bf16* p = P + (size_t)row * SEQ + tid * 8;
#pragma unroll
  for (int i = 0; i < 8; ++i) p[i] = (__bf16)(x[i] * inv);
}

// --- GEMM 3: O = P @ V  (B operand = V^T, K-contiguous), bf16 out -----------
__global__ void __launch_bounds__(256)
gemm_pv_kernel(const __bf16* __restrict__ P, const __bf16* __restrict__ VT,
               __bf16* __restrict__ O) {
  __shared__ __align__(16) __bf16 lds[2 * CBUF];
  const int bz = blockIdx.z;
  const __bf16* Pb = P  + (size_t)bz * SEQ * SEQ;   // [SEQ, SEQ]
  const __bf16* Vb = VT + (size_t)bz * DIM * SEQ;   // [DIM, SEQ]
  __bf16* Ob = O + (size_t)bz * SEQ * DIM;
  const int wave = threadIdx.x >> 5;
  const int lane = threadIdx.x & 31;
  const int m0 = blockIdx.y * 256 + wave * 32;
  const int n0 = blockIdx.x * 64;
  v8f acc[8] = {};
  gemm_compute(Pb, SEQ, Vb, SEQ, DIM, SEQ, m0, n0, lds, acc);

  const int nc = n0 + (lane & 15);
#pragma unroll
  for (int mi = 0; mi < 2; ++mi) {
    const int mb = m0 + 16 * mi + ((lane >> 4) << 3);
#pragma unroll
    for (int r = 0; r < 8; ++r) {
      __bf16* orow = Ob + (size_t)(mb + r) * DIM + nc;
#pragma unroll
      for (int ni = 0; ni < 4; ++ni) orow[16 * ni] = (__bf16)acc[mi * 4 + ni][r];
    }
  }
}

// --- GEMM 4: out = O @ Wout^T + bias, f32 out [ROWS, DIM] -------------------
__global__ void __launch_bounds__(256)
gemm_out_kernel(const __bf16* __restrict__ A, const __bf16* __restrict__ W,
                const float* __restrict__ bias, float* __restrict__ Out) {
  __shared__ __align__(16) __bf16 lds[2 * CBUF];
  const int wave = threadIdx.x >> 5;
  const int lane = threadIdx.x & 31;
  const int m0 = blockIdx.y * 256 + wave * 32;
  const int n0 = blockIdx.x * 64;
  v8f acc[8] = {};
  gemm_compute(A, DIM, W, DIM, DIM, DIM, m0, n0, lds, acc);

  const int nc = n0 + (lane & 15);
  float bi[4];
#pragma unroll
  for (int ni = 0; ni < 4; ++ni) bi[ni] = bias[nc + 16 * ni];
#pragma unroll
  for (int mi = 0; mi < 2; ++mi) {
    const int mb = m0 + 16 * mi + ((lane >> 4) << 3);
#pragma unroll
    for (int r = 0; r < 8; ++r) {
      float* orow = Out + (size_t)(mb + r) * DIM + nc;
#pragma unroll
      for (int ni = 0; ni < 4; ++ni) orow[16 * ni] = acc[mi * 4 + ni][r] + bi[ni];
    }
  }
}

// ---------------------------------------------------------------------------
extern "C" void kernel_launch(void* const* d_in, const int* in_sizes, int n_in,
                              void* d_out, int out_size, void* d_ws, size_t ws_size,
                              hipStream_t stream) {
  (void)in_sizes; (void)n_in; (void)out_size; (void)ws_size;
  const float* x     = (const float*)d_in[0];  // [8, 2048, 1024]
  const float* w_qkv = (const float*)d_in[1];  // [3072, 1024]
  const float* w_out = (const float*)d_in[2];  // [1024, 1024]
  const float* b_out = (const float*)d_in[3];  // [1024]
  float* out = (float*)d_out;                  // [8, 2048, 1024]

  // workspace carve-up (~362 MB)
  __bf16* xb    = (__bf16*)d_ws;                          // ROWS*DIM
  __bf16* wqkvb = xb    + (size_t)ROWS * DIM;             // 3*DIM*DIM
  __bf16* woutb = wqkvb + (size_t)3 * DIM * DIM;          // DIM*DIM
  __bf16* qk    = woutb + (size_t)DIM * DIM;              // ROWS*2*DIM
  __bf16* vt    = qk    + (size_t)ROWS * 2 * DIM;         // BATCH*DIM*SEQ
  float*  S     = (float*)(vt + (size_t)BATCH * DIM * SEQ);   // BATCH*SEQ*SEQ
  __bf16* P     = (__bf16*)(S + (size_t)BATCH * SEQ * SEQ);   // BATCH*SEQ*SEQ
  __bf16* attn  = P + (size_t)BATCH * SEQ * SEQ;              // ROWS*DIM

  cvt_bf16_kernel<<<(ROWS * DIM) / 1024, 256, 0, stream>>>(x, xb, ROWS * DIM);
  cvt_bf16_kernel<<<(3 * DIM * DIM) / 1024, 256, 0, stream>>>(w_qkv, wqkvb,
                                                              3 * DIM * DIM);
  cvt_bf16_kernel<<<(DIM * DIM) / 1024, 256, 0, stream>>>(w_out, woutb,
                                                          DIM * DIM);

  gemm_qkv_kernel<<<dim3((3 * DIM) / 64, ROWS / 256), 256, 0, stream>>>(
      xb, wqkvb, qk, vt);

  gemm_scores_kernel<<<dim3(SEQ / 64, SEQ / 256, BATCH), 256, 0, stream>>>(qk, S);

  softmax_kernel<<<ROWS, 256, 0, stream>>>(S, P);

  gemm_pv_kernel<<<dim3(DIM / 64, SEQ / 256, BATCH), 256, 0, stream>>>(P, vt, attn);

  gemm_out_kernel<<<dim3(DIM / 64, ROWS / 256), 256, 0, stream>>>(attn, woutb,
                                                                  b_out, out);
}
